// MultiHeadAttention_42090679501554
// MI455X (gfx1250) — matrix operand order repaired and checked by reading, not compiled, not run on hardware
//
#include <hip/hip_runtime.h>
#include <hip/hip_bf16.h>
#include <math.h>

// MHA forward: B=4 S=2048 D=1024 H=16 DH=64, fp32 in/out, f16 WMMA internally.
// Cooperative LDS staging via gfx1250 async global->LDS + double buffering.
// GEMM waves compute 32x64 tiles (8 WMMA per k-step) to amortize barriers.

#define Bq 4
#define Sq 2048
#define Dq 1024
#define Hq 16
#define DHq 64

typedef __attribute__((ext_vector_type(16))) _Float16 v16h;
typedef __attribute__((ext_vector_type(8)))  _Float16 v8h;
typedef __attribute__((ext_vector_type(4)))  _Float16 v4h;
typedef __attribute__((ext_vector_type(8)))  float    v8f;
typedef __attribute__((ext_vector_type(4)))  float    v4f;
typedef __attribute__((ext_vector_type(4)))  int      v4i;

#define AS1 __attribute__((address_space(1)))
#define AS3 __attribute__((address_space(3)))

static __device__ __forceinline__ v8f wmma_f16(v16h a, v16h b, v8f c) {
  return __builtin_amdgcn_wmma_f32_16x16x32_f16(
      /*neg_a=*/false, a, /*neg_b=*/false, b,
      /*c_mod=*/(short)0, c, /*reuse_a=*/false, /*reuse_b=*/false);
}

// Copy 16 bytes global -> LDS per lane. Async (ASYNCcnt) when available.
static __device__ __forceinline__ void async_cp16(const _Float16* g, _Float16* l) {
#if __has_builtin(__builtin_amdgcn_global_load_async_to_lds_b128)
  __builtin_amdgcn_global_load_async_to_lds_b128(
      (AS1 v4i*)(g), (AS3 v4i*)(l), 0, 0);
#else
  *(v8h*)l = *(const v8h*)g;
#endif
}

static __device__ __forceinline__ void async_wait0() {
#if __has_builtin(__builtin_amdgcn_global_load_async_to_lds_b128)
#if __has_builtin(__builtin_amdgcn_s_wait_asynccnt)
  __builtin_amdgcn_s_wait_asynccnt(0);
#else
  asm volatile("s_wait_asynccnt 0" ::: "memory");
#endif
#endif
}

// A fragment (16x32, M x K) from row-major f16 matrix, tile origin p, stride ld.
static __device__ __forceinline__ v16h load_a_g(const _Float16* p, int ld) {
  const int lane = threadIdx.x & 31;
  const int m  = lane & 15;
  const int kh = (lane >> 4) << 3;
  const _Float16* r = p + (size_t)m * ld + kh;
  v8h lo = *(const v8h*)(r);
  v8h hi = *(const v8h*)(r + 16);
  return __builtin_shufflevector(lo, hi, 0,1,2,3,4,5,6,7,8,9,10,11,12,13,14,15);
}

// B fragment (32x16, K x N) from memory holding B^T row-major (N rows of K halves).
static __device__ __forceinline__ v16h load_b_any(const _Float16* p, int ld) {
  const int lane = threadIdx.x & 31;
  const _Float16* r = p + (size_t)(lane & 15) * ld + ((lane >> 4) << 3);
  v8h lo = *(const v8h*)(r);
  v8h hi = *(const v8h*)(r + 16);
  return __builtin_shufflevector(lo, hi, 0,1,2,3,4,5,6,7,8,9,10,11,12,13,14,15);
}

// ---------------------------------------------------------------- convert f32 -> f16 (x4)
__global__ void mha_cvt_f16(const v4f* __restrict__ src, v4h* __restrict__ dst, int n4) {
  int i = blockIdx.x * blockDim.x + threadIdx.x;
  if (i < n4) dst[i] = __builtin_convertvector(src[i], v4h);
}

// ---------------------------------------------------------------- QKV projection
// Block (8 waves): 256x64 tile of Y = X @ W^T; per wave 32x64 (8 WMMA / k-step).
__global__ __launch_bounds__(256) void mha_qkv(const _Float16* __restrict__ xh,
                                               const _Float16* __restrict__ wq,
                                               const _Float16* __restrict__ wk,
                                               const _Float16* __restrict__ wv,
                                               _Float16* __restrict__ Qh,
                                               _Float16* __restrict__ Kh,
                                               _Float16* __restrict__ VTh) {
  __shared__ _Float16 bbuf[2][64 * 32];            // W tile: 64 n-rows x 32 k
  const int tid  = threadIdx.x;
  const int wave = tid >> 5;
  const int mt   = blockIdx.x * 8 + wave;          // 32-row tile of X
  const int n0   = blockIdx.y * 64;
  const int which = blockIdx.z;
  const _Float16* w = (which == 0) ? wq : (which == 1) ? wk : wv;
  const _Float16* a_base0 = xh + (size_t)mt * 32 * Dq;
  const _Float16* a_base1 = a_base0 + (size_t)16 * Dq;

  const int brow = tid >> 2;                       // 0..63
  const int bch  = (tid & 3) << 3;                 // 0,8,16,24
  const _Float16* gB = w + (size_t)(n0 + brow) * Dq + bch;
  _Float16* lB0 = &bbuf[0][brow * 32 + bch];
  _Float16* lB1 = &bbuf[1][brow * 32 + bch];

  async_cp16(gB, lB0);                             // stage k=0
  async_wait0();
  __syncthreads();

  v8f c[2][4] = {};
  v16h a_nxt0 = load_a_g(a_base0, Dq);
  v16h a_nxt1 = load_a_g(a_base1, Dq);
  int p = 0;
  for (int kk = 0; kk < Dq / 32; ++kk) {
    const int k = kk * 32;
    if (kk + 1 < Dq / 32)
      async_cp16(gB + k + 32, p ? lB0 : lB1);      // prefetch next W tile
    const v16h a0 = a_nxt0;
    const v16h a1 = a_nxt1;
    if (kk + 1 < Dq / 32) {
      a_nxt0 = load_a_g(a_base0 + k + 32, Dq);
      a_nxt1 = load_a_g(a_base1 + k + 32, Dq);
    }
#pragma unroll
    for (int t = 0; t < 4; ++t) {
      v16h b = load_b_any(&bbuf[p][16 * t * 32], 32);
      c[0][t] = wmma_f16(a0, b, c[0][t]);
      c[1][t] = wmma_f16(a1, b, c[1][t]);
    }
    async_wait0();
    __syncthreads();
    p ^= 1;
  }

  const int lane = tid & 31;
  const int nn = lane & 15;
  const int g  = lane >> 4;
#pragma unroll
  for (int mm = 0; mm < 2; ++mm)
#pragma unroll
    for (int t = 0; t < 4; ++t) {
      const int n = n0 + 16 * t + nn;              // 0..1023
      const int h = n >> 6, d = n & 63;
#pragma unroll
      for (int r = 0; r < 8; ++r) {
        const int row = mt * 32 + mm * 16 + r + g * 8;   // token 0..8191
        const int b = row >> 11;
        const int s = row & (Sq - 1);
        const _Float16 val = (_Float16)c[mm][t][r];
        const size_t head = (size_t)(b * Hq + h);
        if (which == 2) {
          VTh[(head * DHq + d) * Sq + s] = val;    // V^T: (BH, DH, S)
        } else {
          _Float16* dst = (which == 0) ? Qh : Kh;
          dst[(head * Sq + s) * DHq + d] = val;    // (BH, S, DH)
        }
      }
    }
}

// ---------------------------------------------------------------- flash attention
// Block (8 waves) = 128 queries of one head; K / V^T tiles staged in LDS per 32 keys.
__global__ __launch_bounds__(256) void mha_attn(const _Float16* __restrict__ Qh,
                                                const _Float16* __restrict__ Kh,
                                                const _Float16* __restrict__ VTh,
                                                _Float16* __restrict__ ctx) {
  __shared__ _Float16 kbuf[2][32 * 64];            // keys x DH
  __shared__ _Float16 vbuf[2][64 * 32];            // DH x keys
  __shared__ _Float16 pbuf[8][512];                // per-wave P^T[32][16]
  const int tid  = threadIdx.x;
  const int wave = tid >> 5;
  const int lane = tid & 31;
  _Float16* pt = pbuf[wave];

  const int head = blockIdx.x >> 4;                // b*H + h
  const int q0   = ((blockIdx.x & 15) * 8 + wave) * 16;
  const _Float16* Qhead = Qh  + (size_t)head * Sq * DHq;
  const _Float16* Khead = Kh  + (size_t)head * Sq * DHq;
  const _Float16* Vt    = VTh + (size_t)head * DHq * Sq;

  const v16h aq0 = load_a_g(Qhead + (size_t)q0 * DHq + 0,  DHq);
  const v16h aq1 = load_a_g(Qhead + (size_t)q0 * DHq + 32, DHq);

  const int krow = tid >> 3, kch = (tid & 7) << 3; // K tile: 32 rows x 64 halves
  const int vrow = tid >> 2, vch = (tid & 3) << 3; // V^T tile: 64 rows x 32 halves
  const _Float16* gK = Khead + (size_t)krow * DHq + kch;
  const _Float16* gV = Vt + (size_t)vrow * Sq + vch;

  async_cp16(gK, &kbuf[0][krow * 64 + kch]);       // stage keys [0,32)
  async_cp16(gV, &vbuf[0][vrow * 32 + vch]);
  async_wait0();
  __syncthreads();

  const int m = lane & 15;
  const int g = lane >> 4;
  const float scale = 0.125f;                      // 1/sqrt(DH)

  float mrow[8], lrow[8];
  v8f o[4] = {};
#pragma unroll
  for (int r = 0; r < 8; ++r) { mrow[r] = -3.0e38f; lrow[r] = 0.f; }

  int p = 0;
  for (int tb = 0; tb < Sq; tb += 32) {
    if (tb + 32 < Sq) {                            // prefetch next K/V tiles
      async_cp16(gK + (size_t)(tb + 32) * DHq, &kbuf[p ^ 1][krow * 64 + kch]);
      async_cp16(gV + (tb + 32),               &vbuf[p ^ 1][vrow * 32 + vch]);
    }
    // two 16x16 score tiles from LDS K tile
    v8f sc0 = {}, sc1 = {};
    {
      v16h b00 = load_b_any(&kbuf[p][0 * 64 + 0],  64);
      v16h b01 = load_b_any(&kbuf[p][0 * 64 + 32], 64);
      sc0 = wmma_f16(aq0, b00, sc0);
      sc0 = wmma_f16(aq1, b01, sc0);
      v16h b10 = load_b_any(&kbuf[p][16 * 64 + 0],  64);
      v16h b11 = load_b_any(&kbuf[p][16 * 64 + 32], 64);
      sc1 = wmma_f16(aq0, b10, sc1);
      sc1 = wmma_f16(aq1, b11, sc1);
    }
    // online softmax (row M = r + 8*g spans the 16 lanes of this half-wave)
    float p0[8], p1[8], alpha[8];
#pragma unroll
    for (int r = 0; r < 8; ++r) {
      float x0 = sc0[r] * scale, x1 = sc1[r] * scale;
      float mx = fmaxf(x0, x1);
#pragma unroll
      for (int d = 1; d < 16; d <<= 1) mx = fmaxf(mx, __shfl_xor(mx, d, 32));
      const float mnew = fmaxf(mrow[r], mx);
      const float a_   = __expf(mrow[r] - mnew);
      const float e0   = __expf(x0 - mnew);
      const float e1   = __expf(x1 - mnew);
      float rs = e0 + e1;
#pragma unroll
      for (int d = 1; d < 16; d <<= 1) rs += __shfl_xor(rs, d, 32);
      lrow[r] = lrow[r] * a_ + rs;
      mrow[r] = mnew;
      alpha[r] = a_;
      p0[r] = e0; p1[r] = e1;
    }
#pragma unroll
    for (int t = 0; t < 4; ++t)
#pragma unroll
      for (int r = 0; r < 8; ++r) o[t][r] *= alpha[r];

    // write P^T to this wave's LDS region: PT[key_local][query_row]
    v8h h0, h1;
#pragma unroll
    for (int r = 0; r < 8; ++r) { h0[r] = (_Float16)p0[r]; h1[r] = (_Float16)p1[r]; }
    *(v8h*)(pt + (m)      * 16 + g * 8) = h0;
    *(v8h*)(pt + (16 + m) * 16 + g * 8) = h1;
    asm volatile("s_wait_dscnt 0" ::: "memory");

    // reload P in A-fragment layout (16x32, K = key_local)
    v16h ap;
    const int kh = g * 8;
#pragma unroll
    for (int i = 0; i < 8; ++i) ap[i]     = pt[(kh + i) * 16 + m];
#pragma unroll
    for (int i = 0; i < 8; ++i) ap[i + 8] = pt[(16 + kh + i) * 16 + m];

    // O += P @ V from LDS V^T tile
#pragma unroll
    for (int t = 0; t < 4; ++t) {
      v16h bv = load_b_any(&vbuf[p][16 * t * 32], 32);
      o[t] = wmma_f16(ap, bv, o[t]);
    }
    async_wait0();
    __syncthreads();
    p ^= 1;
  }

  // epilogue: ctx (B,S,D) f16
  const int b = head >> 4, h = head & 15;
#pragma unroll
  for (int t = 0; t < 4; ++t)
#pragma unroll
    for (int r = 0; r < 8; ++r) {
      const int s = q0 + r + g * 8;
      const float val = o[t][r] / lrow[r];
      ctx[((size_t)(b * Sq + s)) * Dq + h * DHq + t * 16 + m] = (_Float16)val;
    }
}

// ---------------------------------------------------------------- output projection
__global__ __launch_bounds__(256) void mha_oproj(const _Float16* __restrict__ ctx,
                                                 const _Float16* __restrict__ woh,
                                                 float* __restrict__ out) {
  __shared__ _Float16 bbuf[2][64 * 32];
  const int tid  = threadIdx.x;
  const int wave = tid >> 5;
  const int mt   = blockIdx.x * 8 + wave;          // 32-row tile
  const int n0   = blockIdx.y * 64;
  const _Float16* a_base0 = ctx + (size_t)mt * 32 * Dq;
  const _Float16* a_base1 = a_base0 + (size_t)16 * Dq;

  const int brow = tid >> 2;
  const int bch  = (tid & 3) << 3;
  const _Float16* gB = woh + (size_t)(n0 + brow) * Dq + bch;
  _Float16* lB0 = &bbuf[0][brow * 32 + bch];
  _Float16* lB1 = &bbuf[1][brow * 32 + bch];

  async_cp16(gB, lB0);
  async_wait0();
  __syncthreads();

  v8f c[2][4] = {};
  v16h a_nxt0 = load_a_g(a_base0, Dq);
  v16h a_nxt1 = load_a_g(a_base1, Dq);
  int p = 0;
  for (int kk = 0; kk < Dq / 32; ++kk) {
    const int k = kk * 32;
    if (kk + 1 < Dq / 32)
      async_cp16(gB + k + 32, p ? lB0 : lB1);
    const v16h a0 = a_nxt0;
    const v16h a1 = a_nxt1;
    if (kk + 1 < Dq / 32) {
      a_nxt0 = load_a_g(a_base0 + k + 32, Dq);
      a_nxt1 = load_a_g(a_base1 + k + 32, Dq);
    }
#pragma unroll
    for (int t = 0; t < 4; ++t) {
      v16h b = load_b_any(&bbuf[p][16 * t * 32], 32);
      c[0][t] = wmma_f16(a0, b, c[0][t]);
      c[1][t] = wmma_f16(a1, b, c[1][t]);
    }
    async_wait0();
    __syncthreads();
    p ^= 1;
  }

  const int lane = tid & 31;
  const int nn = lane & 15;
  const int g  = lane >> 4;
#pragma unroll
  for (int mm = 0; mm < 2; ++mm)
#pragma unroll
    for (int t = 0; t < 4; ++t)
#pragma unroll
      for (int r = 0; r < 8; ++r)
        out[(size_t)(mt * 32 + mm * 16 + r + g * 8) * Dq + n0 + 16 * t + nn] =
            c[mm][t][r];
}

// ---------------------------------------------------------------- launcher
extern "C" void kernel_launch(void* const* d_in, const int* in_sizes, int n_in,
                              void* d_out, int out_size, void* d_ws, size_t ws_size,
                              hipStream_t stream) {
  (void)in_sizes; (void)n_in; (void)out_size; (void)ws_size;
  const float* x  = (const float*)d_in[0];
  const float* WQ = (const float*)d_in[1];
  const float* WK = (const float*)d_in[2];
  const float* WV = (const float*)d_in[3];
  const float* WO = (const float*)d_in[4];
  float* out = (float*)d_out;

  _Float16* ws = (_Float16*)d_ws;
  const size_t NX = (size_t)Bq * Sq * Dq;          // 8,388,608
  const size_t NW = (size_t)Dq * Dq;               // 1,048,576
  _Float16* xh  = ws;                              // reused as ctx after QKV
  _Float16* wqh = ws + NX;
  _Float16* wkh = wqh + NW;
  _Float16* wvh = wkh + NW;
  _Float16* woh = wvh + NW;
  _Float16* Qh  = woh + NW;
  _Float16* Kh  = Qh + NX;
  _Float16* VTh = Kh + NX;
  _Float16* ctx = xh;

  mha_cvt_f16<<<dim3((unsigned)(NX / 4 / 256)), 256, 0, stream>>>(
      (const v4f*)x, (v4h*)xh, (int)(NX / 4));
  mha_cvt_f16<<<dim3((unsigned)(NW / 4 / 256)), 256, 0, stream>>>(
      (const v4f*)WQ, (v4h*)wqh, (int)(NW / 4));
  mha_cvt_f16<<<dim3((unsigned)(NW / 4 / 256)), 256, 0, stream>>>(
      (const v4f*)WK, (v4h*)wkh, (int)(NW / 4));
  mha_cvt_f16<<<dim3((unsigned)(NW / 4 / 256)), 256, 0, stream>>>(
      (const v4f*)WV, (v4h*)wvh, (int)(NW / 4));
  mha_cvt_f16<<<dim3((unsigned)(NW / 4 / 256)), 256, 0, stream>>>(
      (const v4f*)WO, (v4h*)woh, (int)(NW / 4));

  mha_qkv<<<dim3(32, 16, 3), 256, 0, stream>>>(xh, wqh, wkh, wvh, Qh, Kh, VTh);
  mha_attn<<<dim3(1024), 256, 0, stream>>>(Qh, Kh, VTh, ctx);
  mha_oproj<<<dim3(32, 16), 256, 0, stream>>>(ctx, woh, out);
}
